// AgentGNN_56513179681278
// MI455X (gfx1250) — matrix operand order repaired and checked by reading, not compiled, not run on hardware
//
#include <hip/hip_runtime.h>
#include <hip/hip_bf16.h>
#include <stdint.h>

// ---------------------------------------------------------------------------
// GATv2 actor/critic GNN for MI455X (gfx1250, wave32).
//  - Dense transforms: WMMA bf16 (v_wmma_f32_16x16x32_bf16), fp32 accumulate.
//    Weights pre-converted to transposed bf16; both A and B tiles staged into
//    LDS by the Tensor Data Mover (tensor_load_to_lds + s_wait_tensorcnt).
//  - Edge phase: wave-per-edge float4 gathers + float atomics (BW-bound part).
// ---------------------------------------------------------------------------

#define NN 20000
#define NE 320000
#define NB 64
#define CH 128

typedef __attribute__((ext_vector_type(16))) __bf16 v16bf;
typedef __attribute__((ext_vector_type(8)))  __bf16 v8bf;
typedef __attribute__((ext_vector_type(8)))  float  v8f;
typedef __attribute__((ext_vector_type(4)))  float  v4f;
typedef __attribute__((ext_vector_type(4)))  unsigned int v4u;
typedef __attribute__((ext_vector_type(8)))  int    v8i;
typedef __attribute__((ext_vector_type(4)))  int    v4i;

#if defined(__gfx1250__) && __has_builtin(__builtin_amdgcn_tensor_load_to_lds) && \
    __has_builtin(__builtin_amdgcn_s_wait_tensorcnt)
#define USE_TDM 1
#else
#define USE_TDM 0
#endif

// ---------------- utility kernels ----------------

__global__ void k_fill(float* p, float v, int n) {
  int i = blockIdx.x * blockDim.x + threadIdx.x;
  if (i < n) p[i] = v;
}

// zero-pad [n,K] -> [n,32]
__global__ void k_pad32(const float* __restrict__ in, float* __restrict__ out, int n, int K) {
  int i = blockIdx.x * blockDim.x + threadIdx.x;
  if (i >= n * 32) return;
  int r = i >> 5, j = i & 31;
  out[i] = (j < K) ? in[(size_t)r * K + j] : 0.f;
}

// BT[col*K + k] = bf16(W[k*128 + col]), zero-padded for k >= KW
__global__ void k_prepW(const float* __restrict__ W, __bf16* __restrict__ BT,
                        int K, int KW) {
  int i = blockIdx.x * blockDim.x + threadIdx.x;
  if (i >= 128 * K) return;
  int col = i / K, k = i % K;
  float v = (k < KW) ? W[(size_t)k * CH + col] : 0.f;
  BT[i] = (__bf16)v;
}

// mean of edge_attr column d over E rows (one block per column)
__global__ void k_mean(const float* __restrict__ ea, int E, int ED, float* __restrict__ mean) {
  __shared__ float red[256];
  int d = blockIdx.x;
  float s = 0.f;
  for (int e = threadIdx.x; e < E; e += 256) s += ea[(size_t)e * ED + d];
  red[threadIdx.x] = s;
  __syncthreads();
  for (int off = 128; off; off >>= 1) {
    if (threadIdx.x < off) red[threadIdx.x] += red[threadIdx.x + off];
    __syncthreads();
  }
  if (threadIdx.x == 0) mean[d] = red[0] / (float)E;
}

// loop_ew[c] = mean_attr . We[:,c]
__global__ void k_loop_ew(const float* __restrict__ mean, const float* __restrict__ We,
                          int ED, float* __restrict__ loop_ew) {
  int c = threadIdx.x;
  float s = 0.f;
  for (int d = 0; d < ED; ++d) s += mean[d] * We[d * CH + c];
  loop_ew[c] = s;
}

// ---- TDM helper: issue one tensor_load_to_lds for a 2D tile (uniform args) --
__device__ __forceinline__ void tdm_load_2d(unsigned lds_off, unsigned long long ga,
                                            unsigned data_size_code, unsigned tdim0,
                                            unsigned tdim1, unsigned tile0, unsigned tile1,
                                            unsigned stride0) {
#if USE_TDM
  v4u g0;
  g0[0] = 1u;                                   // count=1 (valid descriptor)
  g0[1] = lds_off;                              // lds_addr (bytes)
  g0[2] = (unsigned)ga;                         // global_addr[31:0]
  g0[3] = (unsigned)(ga >> 32) | (2u << 30);    // global_addr[56:32] | type=2
  v8i g1;
  g1[0] = (int)(data_size_code << 16);          // data_size, no flags/mask
  g1[1] = (int)((tdim0 & 0xFFFFu) << 16);       // tensor_dim0 lo16
  g1[2] = (int)((tdim0 >> 16) | ((tdim1 & 0xFFFFu) << 16));
  g1[3] = (int)((tdim1 >> 16) | (tile0 << 16)); // tile_dim0
  g1[4] = (int)tile1;                           // tile_dim1, tile_dim2=0
  g1[5] = (int)stride0;                         // tensor_dim0_stride lo32
  g1[6] = 0;
  g1[7] = 0;
  v4i g2 = {0, 0, 0, 0}, g3 = {0, 0, 0, 0};
#if __clang_major__ >= 23
  v8i g4 = {0, 0, 0, 0, 0, 0, 0, 0};
  __builtin_amdgcn_tensor_load_to_lds(g0, g1, g2, g3, g4, 0);
#else
  __builtin_amdgcn_tensor_load_to_lds(g0, g1, g2, g3, 0);
#endif
#else
  (void)lds_off; (void)ga; (void)data_size_code; (void)tdim0; (void)tdim1;
  (void)tile0; (void)tile1; (void)stride0;
#endif
}

// ---------------- WMMA GEMM:  C[M,128] = A[M,K] @ BT^T + bias --------------
// A: fp32, row stride = K = KT*32 (zero-padded).  BT: bf16 [128 cols][K].
// M % 32 == 0. Block = 32 rows x 128 cols, 8 waves; 2 row tiles per wave.
template <int KT>
__global__ __launch_bounds__(256) void k_gemm(const float* __restrict__ A,
                                              const __bf16* __restrict__ BT,
                                              const float* __restrict__ bias,
                                              float* __restrict__ C,
                                              int M, int relu) {
  constexpr int K = KT * 32;
  __shared__ __attribute__((aligned(16))) float  As32[32 * 32];  // 4 KB
  __shared__ __attribute__((aligned(16))) __bf16 Bs[128 * K];    // up to 32 KB
  const int tid  = threadIdx.x;
  const int wave = tid >> 5;         // 0..7 -> column tile
  const int lane = tid & 31;
  const int g    = lane >> 4;        // lane half (fragment layout)
  const int lm   = lane & 15;
  const int rowBase = blockIdx.x * 32;

  v8f acc0 = {}, acc1 = {};

  // ---- preload the whole BT (bf16, already transposed) into LDS, once ----
#if USE_TDM
  if (tid < 32) {
    tdm_load_2d((unsigned)(uintptr_t)&Bs[0], (unsigned long long)(uintptr_t)BT,
                /*2B*/ 1u, (unsigned)K, 128u, (unsigned)K, 128u, (unsigned)K);
  }
#else
  for (int idx = tid * 8; idx < 128 * K; idx += 256 * 8)
    *(v8bf*)&Bs[idx] = *(const v8bf*)&BT[idx];
#endif

#pragma unroll
  for (int kt = 0; kt < KT; ++kt) {
    const int k0 = kt * 32;
    // ---- stage A tile (32 rows x 32 K fp32) ----
#if USE_TDM
    if (tid < 32) {  // wave 0 drives the Tensor Data Mover
      unsigned long long ga =
          (unsigned long long)(uintptr_t)A + ((size_t)rowBase * K + k0) * 4ull;
      tdm_load_2d((unsigned)(uintptr_t)&As32[0], ga, /*4B*/ 2u,
                  (unsigned)K, (unsigned)M, 32u, 32u, (unsigned)K);
      __builtin_amdgcn_s_wait_tensorcnt(0);   // covers B (first iter) + A
    }
#else
    {  // manual fallback: thread t loads one float4
      int idx = tid * 4;
      int r = idx >> 5, kk = idx & 31;
      v4f v = *(const v4f*)&A[(size_t)(rowBase + r) * K + k0 + kk];
      *(v4f*)&As32[r * 32 + kk] = v;
    }
#endif
    __syncthreads();

    // ---- B fragment: k = g*8..+7 and 16+g*8..+7 per lane -> 2x ds_load_b128 ----
    const __bf16* bc = &Bs[(wave * 16 + lm) * K + k0 + g * 8];
    v8bf b0 = *(const v8bf*)bc;
    v8bf b1 = *(const v8bf*)(bc + 16);
    v16bf fbv = __builtin_shufflevector(b0, b1, 0, 1, 2, 3, 4, 5, 6, 7,
                                        8, 9, 10, 11, 12, 13, 14, 15);
    // ---- A fragments (2 row tiles), contiguous fp32 runs -> ds_load_b128 ----
#pragma unroll
    for (int rt = 0; rt < 2; ++rt) {
      const float* ar = &As32[(rt * 16 + lm) * 32];
      v4f a0 = *(const v4f*)(ar + g * 8);
      v4f a1 = *(const v4f*)(ar + g * 8 + 4);
      v4f a2 = *(const v4f*)(ar + 16 + g * 8);
      v4f a3 = *(const v4f*)(ar + 16 + g * 8 + 4);
      v16bf fa;
#pragma unroll
      for (int j = 0; j < 4; ++j) {
        fa[j]      = (__bf16)a0[j];
        fa[4 + j]  = (__bf16)a1[j];
        fa[8 + j]  = (__bf16)a2[j];
        fa[12 + j] = (__bf16)a3[j];
      }
      if (rt == 0)
        acc0 = __builtin_amdgcn_wmma_f32_16x16x32_bf16(false, fa, false, fbv,
                                                       (short)0, acc0, false, false);
      else
        acc1 = __builtin_amdgcn_wmma_f32_16x16x32_bf16(false, fa, false, fbv,
                                                       (short)0, acc1, false, false);
    }
    __syncthreads();
  }

  // ---- store: vgpr r -> row = rt*16 + g*8 + r, col = wave*16 + lane%16 ----
  int col = wave * 16 + lm;
  float bc2 = bias ? bias[col] : 0.f;
#pragma unroll
  for (int r = 0; r < 8; ++r) {
    float v0 = acc0[r] + bc2;
    float v1 = acc1[r] + bc2;
    if (relu) { v0 = fmaxf(v0, 0.f); v1 = fmaxf(v1, 0.f); }
    C[(size_t)(rowBase + (g << 3) + r) * CH + col]      = v0;
    C[(size_t)(rowBase + 16 + (g << 3) + r) * CH + col] = v1;
  }
}

// out[i] = X[i,:] . W + b  (128->1 linears), one wave per row, float4 loads
__global__ void k_lin1(const float* __restrict__ X, const float* __restrict__ W,
                       const float* __restrict__ b, float* __restrict__ out, int M) {
  int row  = (blockIdx.x * blockDim.x + threadIdx.x) >> 5;
  int lane = threadIdx.x & 31;
  if (row >= M) return;
  v4f x = *(const v4f*)&X[(size_t)row * CH + lane * 4];
  v4f w = *(const v4f*)&W[lane * 4];
  float s = x[0] * w[0] + x[1] * w[1] + x[2] * w[2] + x[3] * w[3];
#pragma unroll
  for (int off = 16; off; off >>= 1) s += __shfl_xor(s, off, 32);
  if (lane == 0) out[row] = s + b[0];
}

// ---------------- edge phase ----------------

__device__ inline void atomicMaxF(float* addr, float val) {
  if (val >= 0.f)
    atomicMax((int*)addr, __float_as_int(val));
  else
    atomicMin((unsigned int*)addr, (unsigned int)__float_as_int(val));
}

__device__ inline float lrelu02(float m) { return m > 0.f ? m : 0.2f * m; }

// alpha[e] = leaky_relu(xl[src]+xr[dst]+ea@We) . att ; segment max into amax
__global__ void k_edge_alpha(const int* __restrict__ src, const int* __restrict__ dst,
                             const float* __restrict__ ea, const float* __restrict__ We,
                             const float* __restrict__ loop_ew, const float* __restrict__ att,
                             const float* __restrict__ xl, const float* __restrict__ xr,
                             float* __restrict__ alpha, float* __restrict__ amax,
                             int E, int Nn, int ED) {
  int e    = (blockIdx.x * blockDim.x + threadIdx.x) >> 5;
  int lane = threadIdx.x & 31;
  if (e >= E + Nn) return;
  int s, d;
  if (e < E) { s = src[e]; d = dst[e]; } else { s = d = e - E; }
  int c0 = lane * 4;
  v4f xlv = *(const v4f*)&xl[(size_t)s * CH + c0];
  v4f xrv = *(const v4f*)&xr[(size_t)d * CH + c0];
  v4f av  = *(const v4f*)&att[c0];
  v4f ew  = {0.f, 0.f, 0.f, 0.f};
  if (e < E) {
    float myea = (lane < ED) ? ea[(size_t)e * ED + lane] : 0.f;
    for (int dd = 0; dd < ED; ++dd) {
      float eav = __shfl(myea, dd, 32);
      v4f wv = *(const v4f*)&We[dd * CH + c0];
      ew[0] += eav * wv[0]; ew[1] += eav * wv[1];
      ew[2] += eav * wv[2]; ew[3] += eav * wv[3];
    }
  } else {
    ew = *(const v4f*)&loop_ew[c0];
  }
  float partial = lrelu02(xlv[0] + xrv[0] + ew[0]) * av[0]
                + lrelu02(xlv[1] + xrv[1] + ew[1]) * av[1]
                + lrelu02(xlv[2] + xrv[2] + ew[2]) * av[2]
                + lrelu02(xlv[3] + xrv[3] + ew[3]) * av[3];
#pragma unroll
  for (int off = 16; off; off >>= 1) partial += __shfl_xor(partial, off, 32);
  if (lane == 0) {
    alpha[e] = partial;
    atomicMaxF(&amax[d], partial);
  }
}

// alpha[e] = exp(alpha[e]-amax[dst]); denom[dst] += alpha[e]
__global__ void k_edge_expsum(const int* __restrict__ dst, float* __restrict__ alpha,
                              const float* __restrict__ amax, float* __restrict__ denom,
                              int E, int Nn) {
  int e = blockIdx.x * blockDim.x + threadIdx.x;
  if (e >= E + Nn) return;
  int d = (e < E) ? dst[e] : e - E;
  float w = __expf(alpha[e] - amax[d]);
  alpha[e] = w;
  atomicAdd(&denom[d], w);
}

// out[dst] += xl[src] * alpha[e]/denom[dst]
__global__ void k_edge_aggr(const int* __restrict__ src, const int* __restrict__ dst,
                            const float* __restrict__ xl, const float* __restrict__ alpha,
                            const float* __restrict__ denom, float* __restrict__ out,
                            int E, int Nn) {
  int e    = (blockIdx.x * blockDim.x + threadIdx.x) >> 5;
  int lane = threadIdx.x & 31;
  if (e >= E + Nn) return;
  int s, d;
  if (e < E) { s = src[e]; d = dst[e]; } else { s = d = e - E; }
  float coeff = alpha[e] / denom[d];
  int c0 = lane * 4;
  v4f xlv = *(const v4f*)&xl[(size_t)s * CH + c0];
  float* o = &out[(size_t)d * CH + c0];
  atomicAdd(o + 0, xlv[0] * coeff);
  atomicAdd(o + 1, xlv[1] * coeff);
  atomicAdd(o + 2, xlv[2] * coeff);
  atomicAdd(o + 3, xlv[3] * coeff);
}

// h = relu(h + bias) elementwise over M*CH
__global__ void k_bias_relu(float* __restrict__ h, const float* __restrict__ b, int n) {
  int i = blockIdx.x * blockDim.x + threadIdx.x;
  if (i < n) h[i] = fmaxf(h[i] + b[i & (CH - 1)], 0.f);
}

// ---------------- attentional pooling ----------------

__global__ void k_node_gmax(const int* __restrict__ batch, const float* __restrict__ gate,
                            float* __restrict__ gmax, int Nn) {
  int i = blockIdx.x * blockDim.x + threadIdx.x;
  if (i < Nn) atomicMaxF(&gmax[batch[i]], gate[i]);
}

__global__ void k_node_expsum(const int* __restrict__ batch, float* __restrict__ gate,
                              const float* __restrict__ gmax, float* __restrict__ den, int Nn) {
  int i = blockIdx.x * blockDim.x + threadIdx.x;
  if (i >= Nn) return;
  int b = batch[i];
  float w = __expf(gate[i] - gmax[b]);
  gate[i] = w;
  atomicAdd(&den[b], w);
}

__global__ void k_node_pool(const int* __restrict__ batch, const float* __restrict__ feats,
                            const float* __restrict__ w, const float* __restrict__ den,
                            float* __restrict__ pooled, int Nn) {
  int i    = (blockIdx.x * blockDim.x + threadIdx.x) >> 5;
  int lane = threadIdx.x & 31;
  if (i >= Nn) return;
  int b = batch[i];
  float coeff = w[i] / den[b];
  int c0 = lane * 4;
  v4f fv = *(const v4f*)&feats[(size_t)i * CH + c0];
  float* o = &pooled[(size_t)b * CH + c0];
  atomicAdd(o + 0, fv[0] * coeff);
  atomicAdd(o + 1, fv[1] * coeff);
  atomicAdd(o + 2, fv[2] * coeff);
  atomicAdd(o + 3, fv[3] * coeff);
}

// ---------------------------------------------------------------------------

extern "C" void kernel_launch(void* const* d_in, const int* in_sizes, int n_in,
                              void* d_out, int out_size, void* d_ws, size_t ws_size,
                              hipStream_t stream) {
  (void)in_sizes; (void)n_in; (void)out_size; (void)ws_size;

  // ---- unpack inputs (leaf order of setup_inputs dict) ----
  int p = 0;
  const float* x_p  = (const float*)d_in[p++];
  const int*   ei_p = (const int*)  d_in[p++];   // [2, NE]
  const float* ea_p = (const float*)d_in[p++];   // [NE, 7]
  const float* x_v  = (const float*)d_in[p++];
  const int*   ei_v = (const int*)  d_in[p++];
  const float* ea_v = (const float*)d_in[p++];   // [NE, 2]
  const int*   batch = (const int*) d_in[p++];

  const float *aWl[5], *abl[5], *aWr[5], *abr[5], *aWe[5], *aatt[5], *ab2[5];
  for (int l = 0; l < 5; ++l) {
    aWl[l] = (const float*)d_in[p++]; abl[l] = (const float*)d_in[p++];
    aWr[l] = (const float*)d_in[p++]; abr[l] = (const float*)d_in[p++];
    aWe[l] = (const float*)d_in[p++]; aatt[l] = (const float*)d_in[p++];
    ab2[l] = (const float*)d_in[p++];
  }
  const float* ahW0 = (const float*)d_in[p++]; const float* ahb0 = (const float*)d_in[p++];
  const float* ahW1 = (const float*)d_in[p++]; const float* ahb1 = (const float*)d_in[p++];

  const float *cWl[5], *cbl[5], *cWr[5], *cbr[5], *cWe[5], *catt[5], *cb2[5];
  for (int l = 0; l < 5; ++l) {
    cWl[l] = (const float*)d_in[p++]; cbl[l] = (const float*)d_in[p++];
    cWr[l] = (const float*)d_in[p++]; cbr[l] = (const float*)d_in[p++];
    cWe[l] = (const float*)d_in[p++]; catt[l] = (const float*)d_in[p++];
    cb2[l] = (const float*)d_in[p++];
  }
  const float *gW[3], *gb[3];
  for (int i = 0; i < 3; ++i) { gW[i] = (const float*)d_in[p++]; gb[i] = (const float*)d_in[p++]; }
  const float *pW[2], *pb[2];
  for (int i = 0; i < 2; ++i) { pW[i] = (const float*)d_in[p++]; pb[i] = (const float*)d_in[p++]; }
  const float *fW[3], *fb[3];
  for (int i = 0; i < 3; ++i) { fW[i] = (const float*)d_in[p++]; fb[i] = (const float*)d_in[p++]; }

  // ---- workspace layout (floats) ----
  float* ws = (float*)d_ws;
  size_t o = 0;
  float* hA      = ws + o; o += (size_t)NN * CH;
  float* hB      = ws + o; o += (size_t)NN * CH;
  float* xl      = ws + o; o += (size_t)NN * CH;
  float* xr      = ws + o; o += (size_t)NN * CH;
  float* xp_pad  = ws + o; o += (size_t)NN * 32;
  float* xv_pad  = ws + o; o += (size_t)NN * 32;
  float* alpha   = ws + o; o += (size_t)(NE + NN);
  float* amax    = ws + o; o += NN;
  float* denom   = ws + o; o += NN;
  float* gatebuf = ws + o; o += NN;
  float* gmax    = ws + o; o += NB;
  float* gden    = ws + o; o += NB;
  float* pooled  = ws + o; o += (size_t)NB * CH;
  float* t64a    = ws + o; o += (size_t)NB * CH;
  float* t64b    = ws + o; o += (size_t)NB * CH;
  float* mean_p  = ws + o; o += 8;
  float* mean_v  = ws + o; o += 8;
  float* loopew  = ws + o; o += CH;
  o = (o + 3) & ~(size_t)3;                       // 16B align
  __bf16* BTbuf  = (__bf16*)(ws + o); o += (size_t)CH * CH / 2;  // 128x128 bf16

  float* out_logits = (float*)d_out;        // [NN]
  float* out_values = (float*)d_out + NN;   // [NB]

  const float NEG_INF = -__builtin_huge_valf();
  const int EN = NE + NN;

  // pad narrow inputs to stride-32; mean edge attrs (self-loop fill='mean')
  k_pad32<<<(NN * 32 + 255) / 256, 256, 0, stream>>>(x_p, xp_pad, NN, 17);
  k_pad32<<<(NN * 32 + 255) / 256, 256, 0, stream>>>(x_v, xv_pad, NN, 12);
  k_mean<<<7, 256, 0, stream>>>(ea_p, NE, 7, mean_p);
  k_mean<<<2, 256, 0, stream>>>(ea_v, NE, 2, mean_v);

  // GEMM wrapper: convert W -> transposed bf16 (zero-padded), then WMMA GEMM
  auto gemm = [&](int KT, const float* A, const float* W, int KW, const float* b,
                  float* C, int M, int relu) {
    int K = KT * 32;
    k_prepW<<<(128 * K + 255) / 256, 256, 0, stream>>>(W, BTbuf, K, KW);
    if (KT == 1)
      k_gemm<1><<<M / 32, 256, 0, stream>>>(A, BTbuf, b, C, M, relu);
    else
      k_gemm<4><<<M / 32, 256, 0, stream>>>(A, BTbuf, b, C, M, relu);
  };

  // one GATv2 layer (first=true -> padded 32-wide input)
  auto gat_layer = [&](const float* X, bool first, int KW, const int* ei,
                       const float* ea, int ED, const float* mean_attr,
                       const float* Wl, const float* bl, const float* Wr,
                       const float* br, const float* We, const float* att,
                       const float* bout, float* out) {
    int KT = first ? 1 : 4;
    gemm(KT, X, Wl, KW, bl, xl, NN, 0);
    gemm(KT, X, Wr, KW, br, xr, NN, 0);
    k_loop_ew<<<1, 128, 0, stream>>>(mean_attr, We, ED, loopew);
    k_fill<<<(NN + 255) / 256, 256, 0, stream>>>(amax, NEG_INF, NN);
    k_fill<<<(NN + 255) / 256, 256, 0, stream>>>(denom, 0.f, NN);
    k_fill<<<(NN * CH + 255) / 256, 256, 0, stream>>>(out, 0.f, NN * CH);
    k_edge_alpha<<<(EN + 7) / 8, 256, 0, stream>>>(ei, ei + NE, ea, We, loopew, att,
                                                   xl, xr, alpha, amax, NE, NN, ED);
    k_edge_expsum<<<(EN + 255) / 256, 256, 0, stream>>>(ei + NE, alpha, amax, denom, NE, NN);
    k_edge_aggr<<<(EN + 7) / 8, 256, 0, stream>>>(ei, ei + NE, xl, alpha, denom, out, NE, NN);
    k_bias_relu<<<(NN * CH + 255) / 256, 256, 0, stream>>>(out, bout, NN * CH);
  };

  // ---- actor GNN: 5 GATv2 layers ----
  const float* cur = xp_pad;
  float* bufs[2] = { hA, hB };
  for (int l = 0; l < 5; ++l) {
    float* nxt = bufs[l & 1];
    gat_layer(cur, l == 0, l == 0 ? 17 : CH, ei_p, ea_p, 7, mean_p,
              aWl[l], abl[l], aWr[l], abr[l], aWe[l], aatt[l], ab2[l], nxt);
    cur = nxt;
  }
  // actor head: relu(h@W0+b0) @ W1 + b1 -> logits
  gemm(4, cur, ahW0, CH, ahb0, xl, NN, 1);
  k_lin1<<<(NN + 7) / 8, 256, 0, stream>>>(xl, ahW1, ahb1, out_logits, NN);

  // ---- critic GNN: 5 GATv2 layers ----
  cur = xv_pad;
  for (int l = 0; l < 5; ++l) {
    float* nxt = bufs[l & 1];
    gat_layer(cur, l == 0, l == 0 ? 12 : CH, ei_v, ea_v, 2, mean_v,
              cWl[l], cbl[l], cWr[l], cbr[l], cWe[l], catt[l], cb2[l], nxt);
    cur = nxt;
  }
  // gate_nn: 128->128 relu, 128->128 relu, 128->1
  gemm(4, cur, gW[0], CH, gb[0], xl, NN, 1);
  gemm(4, xl, gW[1], CH, gb[1], xr, NN, 1);
  k_lin1<<<(NN + 7) / 8, 256, 0, stream>>>(xr, gW[2], gb[2], gatebuf, NN);
  // pool_nn: 128->128 relu, 128->128 relu (act_last=True) -> feats in xr
  gemm(4, cur, pW[0], CH, pb[0], xl, NN, 1);
  gemm(4, xl, pW[1], CH, pb[1], xr, NN, 1);
  // per-graph softmax pooling
  k_fill<<<1, 256, 0, stream>>>(gmax, NEG_INF, NB);
  k_fill<<<1, 256, 0, stream>>>(gden, 0.f, NB);
  k_fill<<<(NB * CH + 255) / 256, 256, 0, stream>>>(pooled, 0.f, NB * CH);
  k_node_gmax<<<(NN + 255) / 256, 256, 0, stream>>>(batch, gatebuf, gmax, NN);
  k_node_expsum<<<(NN + 255) / 256, 256, 0, stream>>>(batch, gatebuf, gmax, gden, NN);
  k_node_pool<<<(NN + 7) / 8, 256, 0, stream>>>(batch, xr, gatebuf, gden, pooled, NN);
  // critic_ff: 128->128 relu, 128->128 relu, 128->1
  gemm(4, pooled, fW[0], CH, fb[0], t64a, NB, 1);
  gemm(4, t64a, fW[1], CH, fb[1], t64b, NB, 1);
  k_lin1<<<(NB + 7) / 8, 256, 0, stream>>>(t64b, fW[2], fb[2], out_values, NB);
}
